// InceptionDenseGCN_64484638982694
// MI455X (gfx1250) — matrix operand order repaired and checked by reading, not compiled, not run on hardware
//
#include <hip/hip_runtime.h>

#define B_ 4
#define C_ 64
#define N_ 4096
#define OUT_ 128
#define K_ 20
#define CQ_ 16   // C/4 bottleneck dim
#define C2_ 32   // C/2 graph-feature dim
#define CA_ 96   // 3C/2 concat dim
#define EPS_ 1e-5f
#define TKW_ 8   // waves per knn block

typedef __attribute__((ext_vector_type(2))) float v2f;
typedef __attribute__((ext_vector_type(8))) float v8f;
typedef __attribute__((ext_vector_type(16))) __bf16 v16bf;

// f32 WMMA 16x16x4 (ISA 7.12.2 f32 layouts):
// A: lanes 0-15 -> M=lane, {K0,K1}; lanes 16-31 -> {K2,K3}. B mirrored (lane=N).
// C/D: VGPR j: lanes 0-15 -> (M=j, N=lane); lanes 16-31 -> (M=j+8, N=lane-16).
__device__ __forceinline__ v8f wmma4(v2f a, v2f b, v8f c) {
  return __builtin_amdgcn_wmma_f32_16x16x4_f32(false, a, false, b, (short)0, c,
                                               false, false);
}

// bf16 WMMA 16x16x32, fp32 accumulate.
// A (16x32, 16-bit): lanes 0-15: elems 0-7 = K 0-7, elems 8-15 = K 16-23;
//                    lanes 16-31: K 8-15 and K 24-31.
// B (32x16): column = lane%16; same K order as A (lanes 0-15: K 0-7,16-23; 16-31: K 8-15,24-31).
__device__ __forceinline__ v8f wmmabf(v16bf a, v16bf b, v8f c) {
  return __builtin_amdgcn_wmma_f32_16x16x32_bf16(false, a, false, b, (short)0,
                                                 c, false, false);
}

union BFrag {
  v16bf v;
  uint4 q[2];
};

// ---------- split x into point-major bf16 hi/lo ([B][N][C]) + fp32 sq norms
__global__ void split_kernel(const float* __restrict__ x,
                             float* __restrict__ sq, __bf16* __restrict__ xhi,
                             __bf16* __restrict__ xlo) {
  int t = blockIdx.x * blockDim.x + threadIdx.x;
  if (t >= B_ * N_) return;
  int b = t / N_, p = t % N_;
  const float* xp = x + (size_t)b * C_ * N_ + p;
  __bf16* hp = xhi + (size_t)t * C_;
  __bf16* lp = xlo + (size_t)t * C_;
  float s = 0.f;
#pragma unroll 8
  for (int c = 0; c < C_; ++c) {
    float v = xp[(size_t)c * N_];
    s += v * v;
    __bf16 h = (__bf16)v;            // RNE
    float r = v - (float)h;
    hp[c] = h;
    lp[c] = (__bf16)r;
  }
  sq[t] = s;
}

// -------------------------------------------- residual: r = relu(bn(W_res x))
__global__ __launch_bounds__(32) void res_kernel(
    const float* __restrict__ x, const float* __restrict__ W,
    const float* __restrict__ g, const float* __restrict__ bb,
    const float* __restrict__ mm, const float* __restrict__ vv,
    float* __restrict__ r) {
  const int lane = threadIdx.x;
  const int hi = lane >> 4, li = lane & 15;
  const int n0 = blockIdx.x * 16;
  const int m0 = blockIdx.y * 16;
  const int b = blockIdx.z;
  const float* xb = x + (size_t)b * C_ * N_;
  v8f acc = {};
#pragma unroll
  for (int ks = 0; ks < C_ / 4; ++ks) {
    int kq = 4 * ks + 2 * hi;
    v2f a, bv;
    a.x = W[(m0 + li) * C_ + kq];
    a.y = W[(m0 + li) * C_ + kq + 1];
    bv.x = xb[(size_t)kq * N_ + n0 + li];
    bv.y = xb[(size_t)(kq + 1) * N_ + n0 + li];
    acc = wmma4(a, bv, acc);
  }
#pragma unroll
  for (int j = 0; j < 8; ++j) {
    int ch = m0 + j + hi * 8;
    float s = g[ch] * rsqrtf(vv[ch] + EPS_);
    float t = bb[ch] - mm[ch] * s;
    float y = fmaxf(acc[j] * s + t, 0.f);
    r[((size_t)b * OUT_ + ch) * N_ + n0 + li] = y;
  }
}

// --- KNN: split-bf16 compensated gram (hi*hi + hi*lo + lo*hi, fp32 acc) via
// v_wmma_f32_16x16x32_bf16, fused with per-row top-20 selection.
// Block = 8 waves; wave w sweeps column tiles jt = w, w+8, ...; per-wave lists
// merged across half-waves with shfl_xor(16), then across waves through LDS.
__global__ __launch_bounds__(256) void knn_kernel(
    const __bf16* __restrict__ xhi, const __bf16* __restrict__ xlo,
    const float* __restrict__ sq, int* __restrict__ idx) {
  __shared__ float tile[TKW_][16][17];  // per-wave gram staging (padded)
  __shared__ float md[TKW_][16][K_];    // per-wave merged top-K dists
  __shared__ int mi[TKW_][16][K_];      // per-wave merged top-K indices
  const int lane = threadIdx.x & 31;
  const int wv = threadIdx.x >> 5;
  const int hi = lane >> 4, li = lane & 15;
  const int i0 = blockIdx.x * 16;
  const int b = blockIdx.y;
  const __bf16* xhb = xhi + (size_t)b * N_ * C_;
  const __bf16* xlb = xlo + (size_t)b * N_ * C_;
  const float* sqb = sq + (size_t)b * N_;

  float bd[K_];
  int bi[K_];
#pragma unroll
  for (int t = 0; t < K_; ++t) {
    bd[t] = __builtin_inff();
    bi[t] = 0;
  }
  const float sqRow = sqb[i0 + li];
  const int cbase = hi * 8;

  // A fragments (row tile fixed): point i0+li, two k-steps (K base 0 / 32).
  BFrag Ah[2], Al[2];
#pragma unroll
  for (int kst = 0; kst < 2; ++kst) {
    const __bf16* pa = xhb + (size_t)(i0 + li) * C_ + kst * 32 + hi * 8;
    Ah[kst].q[0] = *(const uint4*)(pa);       // K = kb + {0..7|8..15}
    Ah[kst].q[1] = *(const uint4*)(pa + 16);  // K = kb+16 + {0..7|8..15}
    const __bf16* pl = xlb + (size_t)(i0 + li) * C_ + kst * 32 + hi * 8;
    Al[kst].q[0] = *(const uint4*)(pl);
    Al[kst].q[1] = *(const uint4*)(pl + 16);
  }

  for (int jt = wv; jt < N_ / 16; jt += TKW_) {
    const int j0 = jt * 16;
    if (jt + TKW_ < N_ / 16)  // prefetch next column tile (global_prefetch_b8)
      __builtin_prefetch(xhb + (size_t)(j0 + TKW_ * 16 + li) * C_, 0, 3);
    v8f acc = {};
#pragma unroll
    for (int kst = 0; kst < 2; ++kst) {
      const __bf16* pb = xhb + (size_t)(j0 + li) * C_ + kst * 32 + hi * 8;
      const __bf16* pq = xlb + (size_t)(j0 + li) * C_ + kst * 32 + hi * 8;
      BFrag Bh, Bl;
      Bh.q[0] = *(const uint4*)(pb);       // K = kb + half*8 + 0..7
      Bh.q[1] = *(const uint4*)(pb + 16);  // K = kb+16 + half*8 + 0..7
      Bl.q[0] = *(const uint4*)(pq);
      Bl.q[1] = *(const uint4*)(pq + 16);
      acc = wmmabf(Ah[kst].v, Bh.v, acc);  // hi*hi
      acc = wmmabf(Ah[kst].v, Bl.v, acc);  // hi*lo
      acc = wmmabf(Al[kst].v, Bh.v, acc);  // lo*hi
    }
    // stage gram tile (wave-private buffer; same-wave DS ops are in-order)
#pragma unroll
    for (int j = 0; j < 8; ++j) tile[wv][j + 8 * hi][li] = acc[j];
    __builtin_amdgcn_wave_barrier();
    // lane owns row li, columns cbase..cbase+7 of this tile
#pragma unroll
    for (int cc = 0; cc < 8; ++cc) {
      int col = cbase + cc;
      float d = sqRow + sqb[j0 + col] - 2.f * tile[wv][li][col];
      if (d < bd[K_ - 1]) {
        int jj = j0 + col;
#pragma unroll
        for (int t = 0; t < K_; ++t) {
          if (d < bd[t]) {
            float td = bd[t];
            int ti = bi[t];
            bd[t] = d;
            bi[t] = jj;
            d = td;
            jj = ti;
          }
        }
      }
    }
    __builtin_amdgcn_wave_barrier();
  }
  // merge half-wave lists within the wave (lane pairs l, l^16)
#pragma unroll
  for (int t = 0; t < K_; ++t) {
    float d = __shfl_xor(bd[t], 16, 32);
    int jj = __shfl_xor(bi[t], 16, 32);
#pragma unroll
    for (int u = 0; u < K_; ++u) {
      if (d < bd[u]) {
        float td = bd[u];
        int ti = bi[u];
        bd[u] = d;
        bi[u] = jj;
        d = td;
        jj = ti;
      }
    }
  }
  if (hi == 0) {
#pragma unroll
    for (int t = 0; t < K_; ++t) {
      md[wv][li][t] = bd[t];
      mi[wv][li][t] = bi[t];
    }
  }
  __syncthreads();
  // cross-wave merge: wave 0, lanes 0-15 fold waves 1..7 into their registers
  if (wv == 0 && hi == 0) {
    for (int w = 1; w < TKW_; ++w) {
#pragma unroll
      for (int t = 0; t < K_; ++t) {
        float d = md[w][li][t];
        int jj = mi[w][li][t];
        if (d < bd[K_ - 1]) {
#pragma unroll
          for (int u = 0; u < K_; ++u) {
            if (d < bd[u]) {
              float td = bd[u];
              int ti = bi[u];
              bd[u] = d;
              bi[u] = jj;
              d = td;
              jj = ti;
            }
          }
        }
      }
    }
    int* op = idx + ((size_t)b * N_ + i0 + li) * K_;
#pragma unroll
    for (int t = 0; t < K_; ++t) op[t] = bi[t];
  }
}

// ------------------ bottleneck: h = relu(bn(W_bot x)), stored point-major
__global__ void bot_kernel(const float* __restrict__ x,
                           const float* __restrict__ W,
                           const float* __restrict__ g,
                           const float* __restrict__ bb,
                           const float* __restrict__ mm,
                           const float* __restrict__ vv,
                           float* __restrict__ h) {
  int t = blockIdx.x * blockDim.x + threadIdx.x;
  if (t >= B_ * N_) return;
  int b = t / N_, p = t % N_;
  const float* xp = x + (size_t)b * C_ * N_ + p;
  float acc[CQ_];
#pragma unroll
  for (int o = 0; o < CQ_; ++o) acc[o] = 0.f;
  for (int c = 0; c < C_; ++c) {
    float xv = xp[(size_t)c * N_];
#pragma unroll
    for (int o = 0; o < CQ_; ++o) acc[o] += W[o * C_ + c] * xv;
  }
  float* hp = h + (size_t)t * CQ_;
#pragma unroll
  for (int o = 0; o < CQ_; ++o) {
    float s = g[o] * rsqrtf(vv[o] + EPS_);
    float tt = bb[o] - mm[o] * s;
    hp[o] = fmaxf(acc[o] * s + tt, 0.f);
  }
}

// -------- gcn: o1 (all 20 nbrs, W_g1) and o2 (even nbrs, W_g2) fused.
// B column n = [h(neighbor) ; h(center)] (K = 32 = 8 f32 WMMA k-steps).
// Center features for the 16 points are staged into LDS once via the CDNA5
// async-to-LDS path (GLOBAL_LOAD_ASYNC_TO_LDS_B128, tracked by ASYNCcnt).
__global__ __launch_bounds__(32) void gcn_kernel(
    const float* __restrict__ h, const int* __restrict__ idx,
    const float* __restrict__ W1, const float* __restrict__ g1,
    const float* __restrict__ b1, const float* __restrict__ m1,
    const float* __restrict__ v1, const float* __restrict__ W2,
    const float* __restrict__ g2, const float* __restrict__ b2,
    const float* __restrict__ m2, const float* __restrict__ v2_,
    float* __restrict__ allf) {
  __shared__ float shc[16 * CQ_];  // center features: 16 points x 16 ch
  const int lane = threadIdx.x;
  const int hi = lane >> 4, li = lane & 15;
  const int n0 = blockIdx.x * 16;
  const int m0 = blockIdx.y * 16;  // 0 or 16
  const int b = blockIdx.z;

  // stage 1KB of center features: lane copies 32B (2 x b128, offset advances
  // both LDS and global addresses per ISA 15.18.3).
  {
    unsigned long long ga =
        (unsigned long long)(const void*)(h + ((size_t)b * N_ + n0) * CQ_) +
        (unsigned)lane * 32u;
    unsigned int la =
        (unsigned int)(unsigned long long)(const void*)shc +
        (unsigned)lane * 32u;  // low 32 bits of flat shared addr = LDS offset
    asm volatile(
        "global_load_async_to_lds_b128 %0, %1, off\n\t"
        "global_load_async_to_lds_b128 %0, %1, off offset:16\n\t"
        "s_wait_asynccnt 0"
        :
        : "v"(la), "v"(ga)
        : "memory");
  }
  __builtin_amdgcn_wave_barrier();

  v2f a1[8], a2[8];
#pragma unroll
  for (int ks = 0; ks < 8; ++ks) {
    int kq = 4 * ks + 2 * hi;
    a1[ks].x = W1[(m0 + li) * C2_ + kq];
    a1[ks].y = W1[(m0 + li) * C2_ + kq + 1];
    a2[ks].x = W2[(m0 + li) * C2_ + kq];
    a2[ks].y = W2[(m0 + li) * C2_ + kq + 1];
  }
  float s1[8], t1[8], s2[8], t2[8];
#pragma unroll
  for (int j = 0; j < 8; ++j) {
    int ch = m0 + j + hi * 8;
    float s = g1[ch] * rsqrtf(v1[ch] + EPS_);
    s1[j] = s;
    t1[j] = b1[ch] - m1[ch] * s;
    s = g2[ch] * rsqrtf(v2_[ch] + EPS_);
    s2[j] = s;
    t2[j] = b2[ch] - m2[ch] * s;
  }
  const int n = n0 + li;  // this lane-pair's column / point
  const float* hC = &shc[li * CQ_];  // staged center features (LDS)
  const int* ip = idx + ((size_t)b * N_ + n) * K_;
  float acc1[8], acc2[8];
#pragma unroll
  for (int j = 0; j < 8; ++j) {
    acc1[j] = -__builtin_inff();
    acc2[j] = -__builtin_inff();
  }
  for (int kk = 0; kk < K_; ++kk) {
    const int nbr = ip[kk];
    const float* hN = h + ((size_t)b * N_ + nbr) * CQ_;
    const bool do2 = (kk & 1) == 0;  // dilation 2
    v8f c1 = {}, c2 = {};
#pragma unroll
    for (int ks = 0; ks < 8; ++ks) {
      int kq = 4 * ks + 2 * hi;
      v2f bv;
      const float* src = (ks < 4) ? hN : hC;  // k<16: neighbor, k>=16: center
      int off = (ks < 4) ? kq : (kq - 16);
      bv.x = src[off];
      bv.y = src[off + 1];
      c1 = wmma4(a1[ks], bv, c1);
      if (do2) c2 = wmma4(a2[ks], bv, c2);
    }
#pragma unroll
    for (int j = 0; j < 8; ++j) {
      float y = c1[j] * s1[j] + t1[j];
      y = (y > 0.f) ? y : 0.2f * y;
      acc1[j] = fmaxf(acc1[j], y);
      if (do2) {
        float z = c2[j] * s2[j] + t2[j];
        z = (z > 0.f) ? z : 0.2f * z;
        acc2[j] = fmaxf(acc2[j], z);
      }
    }
  }
#pragma unroll
  for (int j = 0; j < 8; ++j) {
    int ch = m0 + j + hi * 8;
    allf[((size_t)b * CA_ + ch) * N_ + n] = acc1[j];
    allf[((size_t)b * CA_ + 32 + ch) * N_ + n] = acc2[j];
  }
}

// ----- gmax over f1: rows 0-15 = max over neighbor h; rows 16-31 = center h
__global__ void gmax_kernel(const float* __restrict__ h,
                            const int* __restrict__ idx,
                            float* __restrict__ allf) {
  int t = blockIdx.x * blockDim.x + threadIdx.x;
  if (t >= B_ * N_ * C2_) return;
  int c = t % C2_;
  int n = (t / C2_) % N_;
  int b = t / (C2_ * N_);
  float v;
  if (c < CQ_) {
    const int* ip = idx + ((size_t)b * N_ + n) * K_;
    v = -__builtin_inff();
    for (int kk = 0; kk < K_; ++kk)
      v = fmaxf(v, h[((size_t)b * N_ + ip[kk]) * CQ_ + c]);
  } else {
    v = h[((size_t)b * N_ + n) * CQ_ + (c - CQ_)];
  }
  allf[((size_t)b * CA_ + 64 + c) * N_ + n] = v;
}

// -------------------- decoder: out = relu(bn(W_dec allf)) + r  (K = 96)
__global__ __launch_bounds__(32) void dec_kernel(
    const float* __restrict__ allf, const float* __restrict__ W,
    const float* __restrict__ g, const float* __restrict__ bb,
    const float* __restrict__ mm, const float* __restrict__ vv,
    const float* __restrict__ r, float* __restrict__ out) {
  const int lane = threadIdx.x;
  const int hi = lane >> 4, li = lane & 15;
  const int n0 = blockIdx.x * 16;
  const int m0 = blockIdx.y * 16;
  const int b = blockIdx.z;
  const float* fb = allf + (size_t)b * CA_ * N_;
  v8f acc = {};
#pragma unroll
  for (int ks = 0; ks < CA_ / 4; ++ks) {
    int kq = 4 * ks + 2 * hi;
    v2f a, bv;
    a.x = W[(m0 + li) * CA_ + kq];
    a.y = W[(m0 + li) * CA_ + kq + 1];
    bv.x = fb[(size_t)kq * N_ + n0 + li];
    bv.y = fb[(size_t)(kq + 1) * N_ + n0 + li];
    acc = wmma4(a, bv, acc);
  }
#pragma unroll
  for (int j = 0; j < 8; ++j) {
    int ch = m0 + j + hi * 8;
    float s = g[ch] * rsqrtf(vv[ch] + EPS_);
    float t = bb[ch] - mm[ch] * s;
    float y = fmaxf(acc[j] * s + t, 0.f);
    size_t o = ((size_t)b * OUT_ + ch) * N_ + n0 + li;
    out[o] = y + r[o];
  }
}

extern "C" void kernel_launch(void* const* d_in, const int* in_sizes, int n_in,
                              void* d_out, int out_size, void* d_ws,
                              size_t ws_size, hipStream_t stream) {
  (void)in_sizes;
  (void)n_in;
  (void)out_size;
  (void)ws_size;
  const float* x = (const float*)d_in[0];
  const float* W_res = (const float*)d_in[1];
  const float* g_res = (const float*)d_in[2];
  const float* b_res = (const float*)d_in[3];
  const float* m_res = (const float*)d_in[4];
  const float* v_res = (const float*)d_in[5];
  const float* W_bot = (const float*)d_in[6];
  const float* g_bot = (const float*)d_in[7];
  const float* b_bot = (const float*)d_in[8];
  const float* m_bot = (const float*)d_in[9];
  const float* v_bot = (const float*)d_in[10];
  const float* W_g1 = (const float*)d_in[11];
  const float* g_g1 = (const float*)d_in[12];
  const float* b_g1 = (const float*)d_in[13];
  const float* m_g1 = (const float*)d_in[14];
  const float* v_g1 = (const float*)d_in[15];
  const float* W_g2 = (const float*)d_in[16];
  const float* g_g2 = (const float*)d_in[17];
  const float* b_g2 = (const float*)d_in[18];
  const float* m_g2 = (const float*)d_in[19];
  const float* v_g2 = (const float*)d_in[20];
  const float* W_dec = (const float*)d_in[21];
  const float* g_dec = (const float*)d_in[22];
  const float* b_dec = (const float*)d_in[23];
  const float* m_dec = (const float*)d_in[24];
  const float* v_dec = (const float*)d_in[25];

  // workspace carve-up (~20.5 MB)
  float* r = (float*)d_ws;                           // B*OUT*N
  float* sq = r + (size_t)B_ * OUT_ * N_;            // B*N
  float* h = sq + (size_t)B_ * N_;                   // B*N*CQ (point-major)
  float* allf = h + (size_t)B_ * N_ * CQ_;           // B*CA*N
  int* idx = (int*)(allf + (size_t)B_ * CA_ * N_);   // B*N*K
  __bf16* xhi = (__bf16*)(idx + (size_t)B_ * N_ * K_);  // B*N*C bf16
  __bf16* xlo = xhi + (size_t)B_ * N_ * C_;             // B*N*C bf16

  split_kernel<<<(B_ * N_ + 255) / 256, 256, 0, stream>>>(x, sq, xhi, xlo);
  res_kernel<<<dim3(N_ / 16, OUT_ / 16, B_), 32, 0, stream>>>(
      x, W_res, g_res, b_res, m_res, v_res, r);
  knn_kernel<<<dim3(N_ / 16, B_), 256, 0, stream>>>(xhi, xlo, sq, idx);
  bot_kernel<<<(B_ * N_ + 255) / 256, 256, 0, stream>>>(x, W_bot, g_bot, b_bot,
                                                        m_bot, v_bot, h);
  gcn_kernel<<<dim3(N_ / 16, 2, B_), 32, 0, stream>>>(
      h, idx, W_g1, g_g1, b_g1, m_g1, v_g1, W_g2, g_g2, b_g2, m_g2, v_g2,
      allf);
  gmax_kernel<<<(B_ * N_ * C2_ + 255) / 256, 256, 0, stream>>>(h, idx, allf);
  dec_kernel<<<dim3(N_ / 16, OUT_ / 16, B_), 32, 0, stream>>>(
      allf, W_dec, g_dec, b_dec, m_dec, v_dec, r, (float*)d_out);
}
